// GCN_27797028339919
// MI455X (gfx1250) — matrix operand order repaired and checked by reading, not compiled, not run on hardware
//
#include <hip/hip_runtime.h>
#include <hip/hip_bf16.h>
#include <math.h>

typedef __attribute__((ext_vector_type(16))) _Float16 v16h;
typedef __attribute__((ext_vector_type(8)))  float    v8f;

// ---------------------------------------------------------------------------
// Degree / normalization kernels
// ---------------------------------------------------------------------------
__global__ void deg_init_kernel(float* deg, int n) {
    int i = blockIdx.x * blockDim.x + threadIdx.x;
    if (i < n) deg[i] = 1.0f;   // self-loop contributes 1 to every node's degree
}

__global__ void deg_accum_kernel(const int* __restrict__ dst, float* deg, int e) {
    int i = blockIdx.x * blockDim.x + threadIdx.x;
    if (i < e) atomicAdd(&deg[dst[i]], 1.0f);
}

__global__ void deg_to_dinv_kernel(float* deg, int n) {
    int i = blockIdx.x * blockDim.x + threadIdx.x;
    if (i < n) deg[i] = rsqrtf(deg[i]);   // deg >= 1 always (self loop)
}

// ---------------------------------------------------------------------------
// Zero fill (float4)
// ---------------------------------------------------------------------------
__global__ void zero4_kernel(float4* p, long n4) {
    long i = (long)blockIdx.x * blockDim.x + threadIdx.x;
    if (i < n4) p[i] = make_float4(0.f, 0.f, 0.f, 0.f);
}

// ---------------------------------------------------------------------------
// Pre-pack W (fp32 row-major [K,128]) into f16 WMMA B-fragments.
// Fragment layout: frag[kc][wave][lane][i] = (f16) W[kc*32 + 16*(lane>>4) + i][wave*16 + (lane&15)]
// One thread per (kc, wave, lane) writes 16 contiguous halfs (32 B).
// ---------------------------------------------------------------------------
__global__ void pack_w_kernel(const float* __restrict__ W, _Float16* __restrict__ out,
                              int K) {
    int t = blockIdx.x * blockDim.x + threadIdx.x;
    int total = (K / 32) * 256;            // (K/32) chunks * 8 waves * 32 lanes
    if (t >= total) return;
    int lane = t & 31;
    int wave = (t >> 5) & 7;
    int kc   = t >> 8;
    int hi   = lane >> 4;
    int lo   = lane & 15;
    int n    = wave * 16 + lo;
    const float* col = W + (size_t)(kc * 32 + hi * 16) * 128 + n;
    _Float16* o = out + (size_t)t * 16;
#pragma unroll
    for (int i = 0; i < 16; ++i) o[i] = (_Float16)col[(size_t)i * 128];
}

// ---------------------------------------------------------------------------
// WMMA GEMM:  Y[M,128] = A[M,K] * W[K,128]   (fp32 A on the fly -> f16,
// B pre-packed f16 fragments, f32 accumulate). grid.x = ceil(M/16),
// blockDim = 256 (8 waves), wave w owns N-tile w. K fully unrolled.
// ---------------------------------------------------------------------------
template <int K>
__global__ void __launch_bounds__(256)
wmma_gemm_kernel(const float* __restrict__ A, const v16h* __restrict__ Wp,
                 float* __restrict__ Y, int M) {
    constexpr int N = 128;
    const int lane  = threadIdx.x & 31;
    const int wave  = threadIdx.x >> 5;      // 0..7 -> N tile
    const int mtile = blockIdx.x;
    const int hi    = lane >> 4;             // half-wave selector
    const int lo    = lane & 15;

    int m = mtile * 16 + lo;
    if (m >= M) m = M - 1;                   // clamp (stores are guarded)
    const int n = wave * 16 + lo;

    const float* arow = A + (size_t)m * K + hi * 16;
    const v16h*  wrow = Wp + wave * 32 + lane;

    v8f c = {};
#pragma unroll
    for (int kc = 0; kc < K / 32; ++kc) {
        v16h a;
        const float* ap = arow + kc * 32;
        // A tile, contiguous layout: half i -> K = 32kc + 16*hi + i
#pragma unroll
        for (int i = 0; i < 8; ++i) a[i]     = (_Float16)ap[i];
#pragma unroll
        for (int i = 0; i < 8; ++i) a[8 + i] = (_Float16)ap[8 + i];
        v16h b = wrow[kc * 256];             // one contiguous 32-B fragment load
        c = __builtin_amdgcn_wmma_f32_16x16x32_f16(
                /*neg_a=*/false, a, /*neg_b=*/false, b,
                /*c_mod=*/(short)0, c, /*reuse_a=*/false, /*reuse_b=*/false);
    }

    // C/D layout: VGPR r -> row (mtile*16 + r + 8*hi), col n
    float* yb = Y + (size_t)(mtile * 16 + 8 * hi) * N + n;
    if (mtile * 16 + 16 <= M) {              // uniform fast path: full tile
#pragma unroll
        for (int r = 0; r < 8; ++r) yb[(size_t)r * N] = c[r];
    } else {
#pragma unroll
        for (int r = 0; r < 8; ++r)
            if (mtile * 16 + 8 * hi + r < M) yb[(size_t)r * N] = c[r];
    }
}

// ---------------------------------------------------------------------------
// Edge scatter: one wave per edge. acc[dst] += dinv[src]*dinv[dst]*xw[src]
// 128 floats per edge: lane handles 4 (float4 gather + 4 f32 atomics -> L2)
// ---------------------------------------------------------------------------
__global__ void __launch_bounds__(256)
scatter_kernel(const int* __restrict__ src, const int* __restrict__ dst,
               const float* __restrict__ dinv, const float* __restrict__ xw,
               float* __restrict__ acc, int e) {
    int wid  = (int)(((long)blockIdx.x * blockDim.x + threadIdx.x) >> 5);
    int lane = threadIdx.x & 31;
    if (wid >= e) return;
    const int s = src[wid];
    const int d = dst[wid];
    const float nrm = dinv[s] * dinv[d];
    float4 v = ((const float4*)(xw + (size_t)s * 128))[lane];
    float* ad = acc + (size_t)d * 128 + lane * 4;
    atomicAdd(ad + 0, nrm * v.x);
    atomicAdd(ad + 1, nrm * v.y);
    atomicAdd(ad + 2, nrm * v.z);
    atomicAdd(ad + 3, nrm * v.w);
}

// ---------------------------------------------------------------------------
// Finalize: h = ELU(acc + dinv^2 * xw + bias)   (self-loop folded in; in place)
// ---------------------------------------------------------------------------
__device__ __forceinline__ float elu1(float v) {
    return v > 0.f ? v : expm1f(v);
}

__global__ void finalize_kernel(float* __restrict__ acc, const float* __restrict__ xw,
                                const float* __restrict__ dinv,
                                const float* __restrict__ bias, int n) {
    long t = (long)blockIdx.x * blockDim.x + threadIdx.x;
    if (t >= (long)n * 32) return;
    int node = (int)(t >> 5);
    int q    = (int)(t & 31);
    float di   = dinv[node];
    float self = di * di;
    float4 a = *(float4*)(acc + (size_t)node * 128 + q * 4);
    float4 w = *(const float4*)(xw + (size_t)node * 128 + q * 4);
    const float* b = bias + q * 4;
    float4 o;
    o.x = elu1(a.x + self * w.x + b[0]);
    o.y = elu1(a.y + self * w.y + b[1]);
    o.z = elu1(a.z + self * w.z + b[2]);
    o.w = elu1(a.w + self * w.w + b[3]);
    *(float4*)(acc + (size_t)node * 128 + q * 4) = o;
}

// ---------------------------------------------------------------------------
// Head: gather roots, concat(h_td, h_bu) @ fc_w + fc_b, log_softmax
// ---------------------------------------------------------------------------
__global__ void head_kernel(const float* __restrict__ h_td, const float* __restrict__ h_bu,
                            const int* __restrict__ root,
                            const float* __restrict__ fcw, const float* __restrict__ fcb,
                            float* __restrict__ out, int nb) {
    int r = blockIdx.x * blockDim.x + threadIdx.x;
    if (r >= nb) return;
    int node = root[r];
    float l0 = fcb[0], l1 = fcb[1], l2 = fcb[2], l3 = fcb[3];
    const float* a = h_td + (size_t)node * 128;
    const float* b = h_bu + (size_t)node * 128;
    for (int f = 0; f < 128; ++f) {
        float va = a[f];
        const float* wr = fcw + (size_t)f * 4;
        l0 += va * wr[0]; l1 += va * wr[1]; l2 += va * wr[2]; l3 += va * wr[3];
    }
    for (int f = 0; f < 128; ++f) {
        float vb = b[f];
        const float* wr = fcw + (size_t)(128 + f) * 4;
        l0 += vb * wr[0]; l1 += vb * wr[1]; l2 += vb * wr[2]; l3 += vb * wr[3];
    }
    float mx  = fmaxf(fmaxf(l0, l1), fmaxf(l2, l3));
    float sum = expf(l0 - mx) + expf(l1 - mx) + expf(l2 - mx) + expf(l3 - mx);
    float lse = mx + logf(sum);
    out[(size_t)r * 4 + 0] = l0 - lse;
    out[(size_t)r * 4 + 1] = l1 - lse;
    out[(size_t)r * 4 + 2] = l2 - lse;
    out[(size_t)r * 4 + 3] = l3 - lse;
}

// ---------------------------------------------------------------------------
// Host orchestration
// ---------------------------------------------------------------------------
static void run_gcn_layer(const float* in, const float* w, const float* bias,
                          const int* src, const int* dst, const float* dinv,
                          float* xw, float* acc, _Float16* wpack,
                          int nn, int kin, int e, hipStream_t stream) {
    // 0) pack W into f16 B-fragments
    int pt = (kin / 32) * 256;
    pack_w_kernel<<<(pt + 255) / 256, 256, 0, stream>>>(w, wpack, kin);
    // 1) GEMM: xw = in @ w
    if (kin == 256)
        wmma_gemm_kernel<256><<<(nn + 15) / 16, 256, 0, stream>>>(
            in, (const v16h*)wpack, xw, nn);
    else
        wmma_gemm_kernel<128><<<(nn + 15) / 16, 256, 0, stream>>>(
            in, (const v16h*)wpack, xw, nn);
    // 2) zero acc
    long n4 = (long)nn * 32;
    zero4_kernel<<<(unsigned)((n4 + 255) / 256), 256, 0, stream>>>((float4*)acc, n4);
    // 3) scatter over edges (self-loop handled in finalize)
    long thr = (long)e * 32;
    scatter_kernel<<<(unsigned)((thr + 255) / 256), 256, 0, stream>>>(src, dst, dinv, xw, acc, e);
    // 4) finalize: acc = ELU(acc + dinv^2*xw + bias)
    long ft = (long)nn * 32;
    finalize_kernel<<<(unsigned)((ft + 255) / 256), 256, 0, stream>>>(acc, xw, dinv, bias, nn);
}

extern "C" void kernel_launch(void* const* d_in, const int* in_sizes, int n_in,
                              void* d_out, int out_size, void* d_ws, size_t ws_size,
                              hipStream_t stream) {
    const float* x       = (const float*)d_in[0];
    const int*   ei      = (const int*)d_in[1];   // [2, E] row-major
    const int*   bu_ei   = (const int*)d_in[2];
    const int*   root    = (const int*)d_in[3];
    const float* w1      = (const float*)d_in[4];
    const float* b1      = (const float*)d_in[5];
    const float* w2      = (const float*)d_in[6];
    const float* b2      = (const float*)d_in[7];
    const float* w3      = (const float*)d_in[8];
    const float* b3      = (const float*)d_in[9];
    const float* w4      = (const float*)d_in[10];
    const float* b4      = (const float*)d_in[11];
    const float* fcw     = (const float*)d_in[12];
    const float* fcb     = (const float*)d_in[13];

    const int e  = in_sizes[1] / 2;          // 1.6M edges
    const int nn = in_sizes[0] / 256;        // 100k nodes
    const int nb = in_sizes[3];              // 2048 roots

    const int* td_src = ei;
    const int* td_dst = ei + e;
    const int* bu_src = bu_ei;
    const int* bu_dst = bu_ei + e;

    // workspace layout (floats)
    float* ws      = (float*)d_ws;
    float* XW      = ws;                             // nn*128
    float* ACC_TD  = XW     + (size_t)nn * 128;      // nn*128
    float* ACC_BU  = ACC_TD + (size_t)nn * 128;      // nn*128
    float* dinv_td = ACC_BU + (size_t)nn * 128;      // nn
    float* dinv_bu = dinv_td + nn;                   // nn
    _Float16* WP   = (_Float16*)(dinv_bu + nn);      // up to 256*128 halfs (64 KB)
    (void)ws_size;

    // --- symmetric normalization terms for both edge sets ---
    int gb_n = (nn + 255) / 256;
    int gb_e = (e + 255) / 256;
    deg_init_kernel<<<gb_n, 256, 0, stream>>>(dinv_td, nn);
    deg_init_kernel<<<gb_n, 256, 0, stream>>>(dinv_bu, nn);
    deg_accum_kernel<<<gb_e, 256, 0, stream>>>(td_dst, dinv_td, e);
    deg_accum_kernel<<<gb_e, 256, 0, stream>>>(bu_dst, dinv_bu, e);
    deg_to_dinv_kernel<<<gb_n, 256, 0, stream>>>(dinv_td, nn);
    deg_to_dinv_kernel<<<gb_n, 256, 0, stream>>>(dinv_bu, nn);

    // --- top-down branch: h = ELU(gcn(x,w1,b1)); h = ELU(gcn(h,w2,b2)) ---
    run_gcn_layer(x,      w1, b1, td_src, td_dst, dinv_td, XW, ACC_TD, WP, nn, 256, e, stream);
    run_gcn_layer(ACC_TD, w2, b2, td_src, td_dst, dinv_td, XW, ACC_TD, WP, nn, 128, e, stream);

    // --- bottom-up branch ---
    run_gcn_layer(x,      w3, b3, bu_src, bu_dst, dinv_bu, XW, ACC_BU, WP, nn, 256, e, stream);
    run_gcn_layer(ACC_BU, w4, b4, bu_src, bu_dst, dinv_bu, XW, ACC_BU, WP, nn, 128, e, stream);

    // --- head: gather roots, FC, log_softmax ---
    head_kernel<<<(nb + 127) / 128, 128, 0, stream>>>(ACC_TD, ACC_BU, root,
                                                      fcw, fcb, (float*)d_out, nb);
}
